// ImprovedGAE_45268955300494
// MI455X (gfx1250) — matrix operand order repaired and checked by reading, not compiled, not run on hardware
//
#include <hip/hip_runtime.h>
#include <hip/hip_bf16.h>

typedef _Float16 v16h __attribute__((ext_vector_type(16)));
typedef float    v8f  __attribute__((ext_vector_type(8)));

#define NN  10000
#define EE  320000
#define IND 128
#define HID 64
#define LAT 32

// ---------------- degree / norm ----------------
__global__ void k_deg_init(float* deg) {
  int i = blockIdx.x * blockDim.x + threadIdx.x;
  if (i < NN) deg[i] = 1.0f;                      // self loop
}

__global__ void k_deg_count(const long long* __restrict__ ei, float* deg) {
  int e = blockIdx.x * blockDim.x + threadIdx.x;
  if (e < EE) atomicAdd(&deg[(int)ei[EE + e]], 1.0f);
}

__global__ void k_dis(const float* __restrict__ deg, float* __restrict__ dis) {
  int i = blockIdx.x * blockDim.x + threadIdx.x;
  if (i < NN) dis[i] = rsqrtf(deg[i]);            // deg >= 1 always
}

// ---------------- dense GEMMs (row-in-LDS, coalesced W reads) ----------------
// xw = x @ W1 ; skip = x @ Ws1 + bs1     (x: [N,128], out dims 64)
__global__ void k_gemm_in(const float* __restrict__ x, const float* __restrict__ W1,
                          const float* __restrict__ Ws1, const float* __restrict__ bs1,
                          float* __restrict__ xw, float* __restrict__ skip) {
  __shared__ float xs[4][IND];
  int ty = threadIdx.x >> 6;          // 4 rows per block
  int c  = threadIdx.x & 63;
  int row = blockIdx.x * 4 + ty;      // grid sized exactly: N/4 blocks
  xs[ty][c]      = x[row * IND + c];
  xs[ty][c + 64] = x[row * IND + c + 64];
  __syncthreads();
  float a0 = 0.f, a1 = bs1[c];
#pragma unroll 4
  for (int k = 0; k < IND; k++) {
    float xv = xs[ty][k];
    a0 += xv * W1[k * HID + c];
    a1 += xv * Ws1[k * HID + c];
  }
  xw[row * HID + c]   = a0;
  skip[row * HID + c] = a1;
}

// xw = h @ W2 ; skip = h @ Ws2 + bs2    (h: [N,64]; skip only 32 cols)
__global__ void k_gemm_h(const float* __restrict__ h, const float* __restrict__ W2,
                         const float* __restrict__ Ws2, const float* __restrict__ bs2,
                         float* __restrict__ xw, float* __restrict__ skip) {
  __shared__ float hs[4][HID];
  int ty = threadIdx.x >> 6;
  int c  = threadIdx.x & 63;
  int row = blockIdx.x * 4 + ty;
  hs[ty][c] = h[row * HID + c];
  __syncthreads();
  float a0 = 0.f, a1 = (c < LAT) ? bs2[c] : 0.f;
#pragma unroll 4
  for (int k = 0; k < HID; k++) {
    float hv = hs[ty][k];
    a0 += hv * W2[k * HID + c];
    if (c < LAT) a1 += hv * Ws2[k * LAT + c];     // wave-uniform branch
  }
  xw[row * HID + c] = a0;
  if (c < LAT) skip[row * LAT + c] = a1;
}

// xw = h @ W3                            (h: [N,64] -> [N,32])
__global__ void k_gemm_z(const float* __restrict__ h, const float* __restrict__ W3,
                         float* __restrict__ xw) {
  __shared__ float hs[8][HID];
  int ty = threadIdx.x >> 5;          // 8 rows per block
  int c  = threadIdx.x & 31;
  int row = blockIdx.x * 8 + ty;
  hs[ty][c]      = h[row * HID + c];
  hs[ty][c + 32] = h[row * HID + c + 32];
  __syncthreads();
  float a0 = 0.f;
#pragma unroll 4
  for (int k = 0; k < HID; k++) a0 += hs[ty][k] * W3[k * LAT + c];
  xw[row * LAT + c] = a0;
}

// ---------------- GCN aggregation ----------------
// agg = b + xw * dis^2   (self loop folded in)
template <int F>
__global__ void k_agg_init(const float* __restrict__ xw, const float* __restrict__ b,
                           const float* __restrict__ dis, float* __restrict__ agg) {
  int idx = blockIdx.x * blockDim.x + threadIdx.x;
  if (idx < NN * F) {
    int i = idx / F, c = idx % F;
    float d = dis[i];
    agg[idx] = b[c] + xw[idx] * d * d;
  }
}

// agg[dst] += xw[src] * dis[src]*dis[dst]
template <int F>
__global__ void k_scatter(const long long* __restrict__ ei, const float* __restrict__ dis,
                          const float* __restrict__ xw, float* agg) {
  int e = blockIdx.x * blockDim.x + threadIdx.x;
  if (e >= EE) return;
  int s = (int)ei[e];
  int d = (int)ei[EE + e];
  float w = dis[s] * dis[d];
  const float4* src4 = (const float4*)(xw + (size_t)s * F);
  float* dst = agg + (size_t)d * F;
#pragma unroll
  for (int c = 0; c < F / 4; c++) {
    float4 v = src4[c];
    atomicAdd(dst + 4 * c + 0, v.x * w);
    atomicAdd(dst + 4 * c + 1, v.y * w);
    atomicAdd(dst + 4 * c + 2, v.z * w);
    atomicAdd(dst + 4 * c + 3, v.w * w);
  }
}

// ---------------- batch norm (relu applied on the fly) ----------------
__global__ void k_bn_stats(const float* __restrict__ agg, float* __restrict__ mu,
                           float* __restrict__ rstd) {
  int c = blockIdx.x;                 // one block per column, F = HID
  float s1 = 0.f, s2 = 0.f;
  for (int i = threadIdx.x; i < NN; i += blockDim.x) {
    float v = agg[(size_t)i * HID + c];
    v = v > 0.f ? v : 0.f;
    s1 += v; s2 += v * v;
  }
  __shared__ float a1[256], a2[256];
  a1[threadIdx.x] = s1; a2[threadIdx.x] = s2;
  __syncthreads();
  for (int o = 128; o > 0; o >>= 1) {
    if (threadIdx.x < o) {
      a1[threadIdx.x] += a1[threadIdx.x + o];
      a2[threadIdx.x] += a2[threadIdx.x + o];
    }
    __syncthreads();
  }
  if (threadIdx.x == 0) {
    float m = a1[0] / (float)NN;
    float v = a2[0] / (float)NN - m * m;
    mu[c] = m;
    rstd[c] = rsqrtf(v + 1e-5f);
  }
}

// h = bn(relu(agg))*g + be (+ skip)   -- skip may alias h (elementwise)
__global__ void k_bn_apply(const float* __restrict__ agg, const float* __restrict__ mu,
                           const float* __restrict__ rstd, const float* __restrict__ g,
                           const float* __restrict__ be, const float* skip, float* h) {
  int idx = blockIdx.x * blockDim.x + threadIdx.x;
  if (idx < NN * HID) {
    int c = idx & (HID - 1);
    float v = agg[idx];
    v = v > 0.f ? v : 0.f;
    float r = (v - mu[c]) * rstd[c] * g[c] + be[c];
    if (skip) r += skip[idx];
    h[idx] = r;
  }
}

// ---------------- z finalize: z = aggz + skip2; emit f32 (out tail) + f16 (decode input)
__global__ void k_zfin(const float* __restrict__ aggz, const float* __restrict__ skip2,
                       float* __restrict__ out_z, _Float16* __restrict__ zh) {
  int idx = blockIdx.x * blockDim.x + threadIdx.x;
  if (idx < NN * LAT) {
    float v = aggz[idx] + skip2[idx];
    out_z[idx] = v;
    zh[idx] = (_Float16)v;
  }
}

// ---------------- decode: adj = sigmoid(z @ z^T) via v_wmma_f32_16x16x32_f16 ----------
// One wave per 16-row tile x 8 column tiles (A fragment reused across the strip).
// Fragment packing follows CDNA5 ISA 7.12.2 16-bit A/B layouts (wave32).
// Sigmoid uses v_exp_f32 + v_rcp_f32 (hardware recip) to stay store-BW bound.
#define ROW_TILES 625          // 10000 / 16
#define COL_CHUNKS 79          // ceil(625 / 8)

__global__ void k_decode(const _Float16* __restrict__ zh, float* __restrict__ out) {
  int wid  = blockIdx.x * (blockDim.x >> 5) + (threadIdx.x >> 5);
  int lane = threadIdx.x & 31;
  if (wid >= ROW_TILES * COL_CHUNKS) return;     // whole-wave uniform exit
  int ti  = wid / COL_CHUNKS;
  int cj0 = (wid % COL_CHUNKS) * 8;

  int m  = lane & 15;
  int hi = lane >> 4;                             // 0: low K group, 1: high K group

  // A fragment: row (ti*16+m); lanes<16 hold K{0..7,16..23}, lanes>=16 hold K{8..15,24..31}
  const _Float16* pa = zh + (size_t)(ti * 16 + m) * LAT + hi * 8;
  union { uint4 u[2]; v16h v; } ua;
  ua.u[0] = *(const uint4*)(pa);
  ua.u[1] = *(const uint4*)(pa + 16);
  v16h afrag = ua.v;

  for (int t = 0; t < 8; t++) {
    int tj = cj0 + t;
    if (tj >= ROW_TILES) break;                   // uniform per wave
    // B fragment: column n = (lane&15) of B = row (tj*16+n) of z;
    // lanes<16 hold K{0..7,16..23}, lanes>=16 hold K{8..15,24..31}
    const _Float16* pb = zh + (size_t)(tj * 16 + m) * LAT + hi * 8;
    union { uint4 u[2]; v16h v; } ub;
    ub.u[0] = *(const uint4*)(pb);
    ub.u[1] = *(const uint4*)(pb + 16);

    v8f c = {};
    c = __builtin_amdgcn_wmma_f32_16x16x32_f16(
        /*neg_a=*/false, afrag, /*neg_b=*/false, ub.v,
        /*c_mod=*/(short)0, c, /*reuse_a=*/false, /*reuse_b=*/false);

    // C/D layout: VGPR r -> row (r + 8*hi), col (lane&15); fuse sigmoid, stream out
#pragma unroll
    for (int r = 0; r < 8; r++) {
      long long row = (long long)(ti * 16 + r + 8 * hi);
      long long col = (long long)(tj * 16 + m);
      // sigmoid(x) = rcp(1 + exp(-x)): v_exp_f32 + v_add_f32 + v_rcp_f32
      float e = __expf(-c[r]);
      float s = __builtin_amdgcn_rcpf(1.0f + e);
      out[row * (long long)NN + col] = s;
    }
  }
}

// ---------------- host orchestration ----------------
extern "C" void kernel_launch(void* const* d_in, const int* in_sizes, int n_in,
                              void* d_out, int out_size, void* d_ws, size_t ws_size,
                              hipStream_t stream) {
  const float*     x   = (const float*)d_in[0];
  const long long* ei  = (const long long*)d_in[1];
  const float* W1  = (const float*)d_in[2];
  const float* b1  = (const float*)d_in[3];
  const float* W2  = (const float*)d_in[4];
  const float* b2  = (const float*)d_in[5];
  const float* W3  = (const float*)d_in[6];
  const float* b3  = (const float*)d_in[7];
  const float* Ws1 = (const float*)d_in[8];
  const float* bs1 = (const float*)d_in[9];
  const float* Ws2 = (const float*)d_in[10];
  const float* bs2 = (const float*)d_in[11];
  const float* g1  = (const float*)d_in[12];
  const float* be1 = (const float*)d_in[13];
  const float* g2  = (const float*)d_in[14];
  const float* be2 = (const float*)d_in[15];
  float* out = (float*)d_out;

  char* wsb = (char*)d_ws;
  size_t off = 0;
  auto carve = [&](size_t bytes) -> void* {
    void* p = wsb + off;
    off += (bytes + 255) & ~(size_t)255;
    return p;
  };
  float* deg  = (float*)carve((size_t)NN * 4);
  float* dis  = (float*)carve((size_t)NN * 4);
  float* buf0 = (float*)carve((size_t)NN * HID * 4);   // xw1 / xw2 / xw3
  float* buf1 = (float*)carve((size_t)NN * HID * 4);   // agg1 / agg2 / aggz
  float* buf2 = (float*)carve((size_t)NN * HID * 4);   // skip1 -> h1 -> h2
  float* buf3 = (float*)carve((size_t)NN * LAT * 4);   // skip2
  float* mu   = (float*)carve(HID * 4);
  float* rstd = (float*)carve(HID * 4);
  _Float16* zh = (_Float16*)carve((size_t)NN * LAT * 2);

  const int B = 256;
  // normalization terms
  k_deg_init<<<(NN + B - 1) / B, B, 0, stream>>>(deg);
  k_deg_count<<<EE / B, B, 0, stream>>>(ei, deg);
  k_dis<<<(NN + B - 1) / B, B, 0, stream>>>(deg, dis);

  // layer 1
  k_gemm_in<<<NN / 4, B, 0, stream>>>(x, W1, Ws1, bs1, buf0, buf2);
  k_agg_init<HID><<<NN * HID / B, B, 0, stream>>>(buf0, b1, dis, buf1);
  k_scatter<HID><<<EE / B, B, 0, stream>>>(ei, dis, buf0, buf1);
  k_bn_stats<<<HID, B, 0, stream>>>(buf1, mu, rstd);
  k_bn_apply<<<NN * HID / B, B, 0, stream>>>(buf1, mu, rstd, g1, be1, buf2, buf2);

  // layer 2
  k_gemm_h<<<NN / 4, B, 0, stream>>>(buf2, W2, Ws2, bs2, buf0, buf3);
  k_agg_init<HID><<<NN * HID / B, B, 0, stream>>>(buf0, b2, dis, buf1);
  k_scatter<HID><<<EE / B, B, 0, stream>>>(ei, dis, buf0, buf1);
  k_bn_stats<<<HID, B, 0, stream>>>(buf1, mu, rstd);
  k_bn_apply<<<NN * HID / B, B, 0, stream>>>(buf1, mu, rstd, g2, be2, nullptr, buf2);

  // layer 3 + skip2
  k_gemm_z<<<NN / 8, B, 0, stream>>>(buf2, W3, buf0);
  k_agg_init<LAT><<<NN * LAT / B, B, 0, stream>>>(buf0, b3, dis, buf1);
  k_scatter<LAT><<<EE / B, B, 0, stream>>>(ei, dis, buf0, buf1);
  k_zfin<<<NN * LAT / B, B, 0, stream>>>(buf1, buf3, out + (size_t)NN * NN, zh);

  // dense decode: sigmoid(z z^T), WMMA f16->f32, 400 MB streamed out
  int waves  = ROW_TILES * COL_CHUNKS;
  int blocks = (waves + 7) / 8;                  // 8 waves (256 threads) per block
  k_decode<<<blocks, B, 0, stream>>>(zh, out);
}
